// lstm_model_20469814132821
// MI455X (gfx1250) — matrix operand order repaired and checked by reading, not compiled, not run on hardware
//
#include <hip/hip_runtime.h>

typedef __attribute__((ext_vector_type(16))) _Float16 v16h;
typedef __attribute__((ext_vector_type(8)))  _Float16 v8h;
typedef __attribute__((ext_vector_type(2)))  _Float16 v2h;
typedef __attribute__((ext_vector_type(8)))  float    v8f;

#define B_TOT      4096
#define T_LEN      200
#define EMB_D      32
#define G1         128   // 4*H1
#define LDS_STRIDE 40    // padded row stride (halves); 80B = multiple of 16B

// ---- fast transcendentals: prefer the CDNA5 v_tanh_f32 unit ----
#if __has_builtin(__builtin_amdgcn_tanhf)
  #define FAST_TANH(x) __builtin_amdgcn_tanhf(x)
  #define HAVE_HW_TANH 1
#elif __has_builtin(__builtin_amdgcn_tanh_f32)
  #define FAST_TANH(x) __builtin_amdgcn_tanh_f32(x)
  #define HAVE_HW_TANH 1
#else
  #define HAVE_HW_TANH 0
#endif

__device__ __forceinline__ float fast_sigmoid(float x) {
#if HAVE_HW_TANH
    return fmaf(0.5f, FAST_TANH(0.5f * x), 0.5f);   // fma + v_tanh_f32 + fma
#else
    return __builtin_amdgcn_rcpf(1.0f + __builtin_amdgcn_exp2f(x * -1.44269504089f));
#endif
}

__device__ __forceinline__ float fast_tanh(float x) {
#if HAVE_HW_TANH
    return FAST_TANH(x);
#else
    return fmaf(2.0f,
                __builtin_amdgcn_rcpf(1.0f + __builtin_amdgcn_exp2f(x * -2.88539008178f)),
                -1.0f);
#endif
}

// z_tile = bias + eA*Wt + hA*Ut  (16x16 f32 accumulator tile)
__device__ __forceinline__ v8f wmma_pair(v16h eA, v16h hA, v16h Wt, v16h Ut, float bias) {
    v8f c;
#pragma unroll
    for (int j = 0; j < 8; ++j) c[j] = bias;
    c = __builtin_amdgcn_wmma_f32_16x16x32_f16(false, eA, false, Wt, (short)0, c, false, false);
    c = __builtin_amdgcn_wmma_f32_16x16x32_f16(false, hA, false, Ut, (short)0, c, false, false);
    return c;
}

__global__ void __launch_bounds__(32, 1)
lstm_fused_kernel(const int*   __restrict__ x,
                  const float* __restrict__ emb,
                  const float* __restrict__ W1,
                  const float* __restrict__ U1,
                  const float* __restrict__ b1,
                  const float* __restrict__ W2,
                  const float* __restrict__ U2,
                  const float* __restrict__ b2,
                  float*       __restrict__ out)
{
    // per-wave (one wave per block) transpose staging buffer: 16 rows x 32 cols f16, padded
    __shared__ __align__(16) _Float16 hbuf[16 * LDS_STRIDE];

    const int  lane    = threadIdx.x;      // 0..31 (wave32)
    const int  nn      = lane & 15;        // N-column within tile / batch row for A loads
    const bool hi      = lane >= 16;
    const int  klo     = hi ? 8 : 0;       // A-operand K chunk base (per ISA 16-bit A layout)
    const int  kb      = hi ? 16 : 0;      // unused; B loads use klo, same K order as A
    const int  rowbase = hi ? 8 : 0;       // D-layout row group for stores
    const int  wb0     = blockIdx.x * 16;  // batch tile start

    // ---- Load B operands for W1 and U1 in WMMA B layout (halves: K=(j<8)?klo+j:16+klo+j-8, N=tile*16+nn) ----
    v16h Wt[8], Ut[8];
#pragma unroll
    for (int n = 0; n < 8; ++n) {
        const int col = n * 16 + nn;
#pragma unroll
        for (int j = 0; j < 16; ++j) {
            Wt[n][j] = (_Float16)W1[((j < 8) ? (klo + j) : (16 + klo + (j - 8))) * G1 + col];
            Ut[n][j] = (_Float16)U1[((j < 8) ? (klo + j) : (16 + klo + (j - 8))) * G1 + col];
        }
    }
    float bias[8];
#pragma unroll
    for (int n = 0; n < 8; ++n) bias[n] = b1[n * 16 + nn];

    // ---- Layer-2 weights replicated in per-lane A-operand layout ----
    v16h w2h[4];
#pragma unroll
    for (int g = 0; g < 4; ++g) {
#pragma unroll
        for (int j = 0; j < 16; ++j) {
            const int k = (j < 8) ? (klo + j) : (16 + klo + (j - 8));
            w2h[g][j] = (_Float16)W2[k * 4 + g];
        }
    }
    float u2[4], bv2[4];
#pragma unroll
    for (int g = 0; g < 4; ++g) { u2[g] = U2[g]; bv2[g] = b2[g]; }

    // ---- state ----
    v16h hA = {};            // layer1 hidden in A layout (t=0: zeros)
    float c1[16];            // layer1 cell in D layout: [tile(2) * 8 + vgpr]
#pragma unroll
    for (int i = 0; i < 16; ++i) c1[i] = 0.f;
    float h2 = 0.f, c2 = 0.f;

    const int xrow = (wb0 + nn) * T_LEN;

    // ---- software-pipelined embedding gather: prefetch step t while computing step t-1 ----
    // preload t = 0
    int idx = x[xrow];
    const float* er0 = emb + (long)idx * EMB_D;
    float4 p0 = *(const float4*)(er0 + klo);
    float4 p1 = *(const float4*)(er0 + klo + 4);
    float4 p2 = *(const float4*)(er0 + klo + 16);
    float4 p3 = *(const float4*)(er0 + klo + 20);

    for (int t = 0; t < T_LEN; ++t) {
        // consume prefetched row -> A layout (lane's K chunks are contiguous within the row)
        v16h eA;
        eA[0]=(_Float16)p0.x; eA[1]=(_Float16)p0.y; eA[2]=(_Float16)p0.z; eA[3]=(_Float16)p0.w;
        eA[4]=(_Float16)p1.x; eA[5]=(_Float16)p1.y; eA[6]=(_Float16)p1.z; eA[7]=(_Float16)p1.w;
        eA[8]=(_Float16)p2.x; eA[9]=(_Float16)p2.y; eA[10]=(_Float16)p2.z; eA[11]=(_Float16)p2.w;
        eA[12]=(_Float16)p3.x; eA[13]=(_Float16)p3.y; eA[14]=(_Float16)p3.z; eA[15]=(_Float16)p3.w;

        // issue next step's gather now (branchless clamp); consumed next iteration, so the
        // x-index load + dependent emb-row loads are fully hidden under this step's compute
        const int tn = (t + 1 < T_LEN) ? (t + 1) : (T_LEN - 1);
        const int idxn = x[xrow + tn];
        const float* ern = emb + (long)idxn * EMB_D;
        p0 = *(const float4*)(ern + klo);
        p1 = *(const float4*)(ern + klo + 4);
        p2 = *(const float4*)(ern + klo + 16);
        p3 = *(const float4*)(ern + klo + 20);

        // ---- layer-1 gates: process the two 16-column halves to limit accumulator pressure ----
#pragma unroll
        for (int nt = 0; nt < 2; ++nt) {
            v8f zi = wmma_pair(eA, hA, Wt[nt],     Ut[nt],     bias[nt]);
            v8f zf = wmma_pair(eA, hA, Wt[2 + nt], Ut[2 + nt], bias[2 + nt]);
            v8f zg = wmma_pair(eA, hA, Wt[4 + nt], Ut[4 + nt], bias[4 + nt]);
            v8f zo = wmma_pair(eA, hA, Wt[6 + nt], Ut[6 + nt], bias[6 + nt]);
#pragma unroll
            for (int v = 0; v < 8; ++v) {
                const float iv = fast_sigmoid(zi[v]);
                const float fv = fast_sigmoid(zf[v]);
                const float gv = fast_tanh(zg[v]);
                const float ov = fast_sigmoid(zo[v]);
                const float cn = fmaf(fv, c1[nt * 8 + v], iv * gv);
                c1[nt * 8 + v] = cn;
                const float hv = ov * fast_tanh(cn);
                // D layout -> LDS row-major f16: row = rowbase+v, col = nt*16+nn
                hbuf[(rowbase + v) * LDS_STRIDE + nt * 16 + nn] = (_Float16)hv;
            }
        }

        // compiler ordering barrier; hardware keeps per-wave DS ops in order
        asm volatile("" ::: "memory");

        // ---- reload h in A layout: lane nn reads row nn, chunks K=[klo,klo+8) and [16+klo,...) ----
        v8h lo8 = *(const v8h*)(hbuf + nn * LDS_STRIDE + klo);
        v8h hi8 = *(const v8h*)(hbuf + nn * LDS_STRIDE + 16 + klo);
        hA = __builtin_shufflevector(lo8, hi8, 0,1,2,3,4,5,6,7,8,9,10,11,12,13,14,15);

        // ---- fused layer 2 (H=1): per-lane half-dot + cross-half shuffle reduce ----
        float acc0 = 0.f, acc1 = 0.f, acc2 = 0.f, acc3 = 0.f;
#if __has_builtin(__builtin_amdgcn_fdot2)
#pragma unroll
        for (int r = 0; r < 8; ++r) {
            v2h hp  = { hA[2 * r], hA[2 * r + 1] };
            v2h w0p = { w2h[0][2 * r], w2h[0][2 * r + 1] };
            v2h w1p = { w2h[1][2 * r], w2h[1][2 * r + 1] };
            v2h w2p = { w2h[2][2 * r], w2h[2][2 * r + 1] };
            v2h w3p = { w2h[3][2 * r], w2h[3][2 * r + 1] };
            acc0 = __builtin_amdgcn_fdot2(hp, w0p, acc0, false);
            acc1 = __builtin_amdgcn_fdot2(hp, w1p, acc1, false);
            acc2 = __builtin_amdgcn_fdot2(hp, w2p, acc2, false);
            acc3 = __builtin_amdgcn_fdot2(hp, w3p, acc3, false);
        }
#else
#pragma unroll
        for (int j = 0; j < 16; ++j) {
            const float hv = (float)hA[j];
            acc0 = fmaf(hv, (float)w2h[0][j], acc0);
            acc1 = fmaf(hv, (float)w2h[1][j], acc1);
            acc2 = fmaf(hv, (float)w2h[2][j], acc2);
            acc3 = fmaf(hv, (float)w2h[3][j], acc3);
        }
#endif
        acc0 += __shfl_xor(acc0, 16, 32);
        acc1 += __shfl_xor(acc1, 16, 32);
        acc2 += __shfl_xor(acc2, 16, 32);
        acc3 += __shfl_xor(acc3, 16, 32);

        const float i2 = fast_sigmoid(fmaf(u2[0], h2, acc0 + bv2[0]));
        const float f2 = fast_sigmoid(fmaf(u2[1], h2, acc1 + bv2[1]));
        const float g2 = fast_sigmoid(fmaf(u2[2], h2, acc2 + bv2[2]));  // layer-2 cell act = sigmoid
        const float o2 = fast_sigmoid(fmaf(u2[3], h2, acc3 + bv2[3]));
        c2 = fmaf(f2, c2, i2 * g2);
        h2 = o2 * fast_sigmoid(c2);                                     // layer-2 output act = sigmoid
    }

    if (!hi) out[wb0 + nn] = h2;   // [B,1] float32, last timestep only
}

extern "C" void kernel_launch(void* const* d_in, const int* in_sizes, int n_in,
                              void* d_out, int out_size, void* d_ws, size_t ws_size,
                              hipStream_t stream) {
    (void)in_sizes; (void)n_in; (void)out_size; (void)d_ws; (void)ws_size;
    const int*   x   = (const int*)  d_in[0];
    const float* emb = (const float*)d_in[1];
    const float* W1  = (const float*)d_in[2];
    const float* U1  = (const float*)d_in[3];
    const float* b1  = (const float*)d_in[4];
    const float* W2  = (const float*)d_in[5];
    const float* U2  = (const float*)d_in[6];
    const float* b2  = (const float*)d_in[7];
    float* out = (float*)d_out;

    dim3 grid(B_TOT / 16);   // 256 waves, one 16-row batch tile each
    dim3 block(32);          // one wave32 per block
    hipLaunchKernelGGL(lstm_fused_kernel, grid, block, 0, stream,
                       x, emb, W1, U1, b1, W2, U2, b2, out);
}
